// CredalAttention_44762149159107
// MI455X (gfx1250) — compile-verified, matrix operand order checked by reading, not run on hardware
//
#include <hip/hip_runtime.h>
#include <hip/hip_bf16.h>

// ---------------------------------------------------------------------------
// Credal multi-head attention for MI455X (gfx1250, wave32, WMMA + TDM).
//   L=S=2048, N=2, E=1024, H=16, d=64  -> 32 attention batches b = n*16+h
// Pipeline (all bf16-operand / f32-accumulate v_wmma_f32_16x16x32_bf16):
//   1) qkv_proj :  Q/K/V = X @ W^T + b   (Q pre-scaled by 1/sqrt(d))
//   2) attn_stats: flash pass over S -> per-row (max m, sumexp)
//   3) uncertainty: avg over heads of S/(sumexp+S+1e-10)
//   4) attn_weights_pv: recompute scores, w=(exp(s-m)+1)/(sumexp+S+1e-10),
//      write the 537MB weights tensor once, fuse P@V via WMMA
//   5) out_proj:  attn_out @ Wo^T + bo  -> f32 output
// Q/K tile staging goes through the Tensor Data Mover (tensor_load_to_lds)
// where the toolchain exposes it; D# descriptor encodes the 64->72-element
// padded LDS pitch natively (pad_interval=32 DWORDs, pad_amount=4 DWORDs).
// ---------------------------------------------------------------------------

typedef __bf16 bf16_t;
typedef __attribute__((ext_vector_type(16))) __bf16    v16bf;
typedef __attribute__((ext_vector_type(8)))  float     v8f;
typedef __attribute__((ext_vector_type(4)))  float     v4f;
typedef __attribute__((ext_vector_type(4)))  unsigned  v4u;
typedef __attribute__((ext_vector_type(2)))  unsigned  v2u;
typedef __attribute__((ext_vector_type(8)))  int       v8i;
typedef __attribute__((ext_vector_type(4)))  int       v4i;

#define LDP 72  // LDS pitch in bf16 elems: 64 + 8 pad (16B aligned rows, conflict break)

#if defined(__has_builtin)
#if __has_builtin(__builtin_amdgcn_tensor_load_to_lds) && \
    __has_builtin(__builtin_amdgcn_s_wait_tensorcnt)
#define HAVE_TDM 1
#endif
#endif
#ifndef HAVE_TDM
#define HAVE_TDM 0
#endif

static __device__ __forceinline__ bf16_t f2bf(float f) {
  union { float f; unsigned u; } v; v.f = f;
  unsigned r = v.u + 0x7FFFu + ((v.u >> 16) & 1u);     // round-to-nearest-even
  union { unsigned short u; bf16_t b; } o; o.u = (unsigned short)(r >> 16);
  return o.b;
}

// pack two f32 -> dword of two bf16
static __device__ __forceinline__ unsigned pk2(float lo, float hi) {
  union { bf16_t b[2]; unsigned u; } o;
  o.b[0] = f2bf(lo); o.b[1] = f2bf(hi);
  return o.u;
}

static __device__ __forceinline__ v8f vzero8() {
  v8f z = {0.f, 0.f, 0.f, 0.f, 0.f, 0.f, 0.f, 0.f};
  return z;
}

// 16x32 bf16 WMMA operand fragment from an LDS tile (row-major, pitch ld).
// CDNA5 ISA 7.12.2 (16-bit A-matrix 16x32):
//   lanes 0-15 : row = lane,    K = {kc+0..7,  kc+16..23}
//   lanes 16-31: row = lane-16, K = {kc+8..15, kc+24..31}
// NT-layout B operand (K x 16) is symmetric with "row" = output column.
static __device__ __forceinline__ v16bf ldfrag(const bf16_t* __restrict__ base, int ld, int kc) {
  int lane = (int)(threadIdx.x & 31u);
  const bf16_t* p = base + (lane & 15) * ld + kc + ((lane >> 4) << 3);
  v16bf f;
#pragma unroll
  for (int i = 0; i < 8; ++i) { f[i] = p[i]; f[i + 8] = p[i + 16]; }
  return f;
}

static __device__ __forceinline__ v8f wmma_bf16(v16bf a, v16bf b, v8f c) {
  // emits v_wmma_f32_16x16x32_bf16
  return __builtin_amdgcn_wmma_f32_16x16x32_bf16(false, a, false, b, (short)0, c, false, false);
}

// ---------------------------------------------------------------------------
// TDM: DMA a rows x 64 bf16 tile (row-major, stride 64) from global into an
// LDS tile of pitch 72 (TDM padding: +4 DWORDs after every 32 DWORDs).
// Descriptor per CDNA5 ISA 8.3/8.4. Issue from one wave; TDM ignores EXEC.
// ---------------------------------------------------------------------------
static __device__ __forceinline__ void tdm_load_tile_bf16(const bf16_t* gsrc,
                                                          bf16_t* lds_dst, int rows) {
#if HAVE_TDM
  unsigned long long ga = (unsigned long long)(uintptr_t)gsrc;
  unsigned lds = (unsigned)(uintptr_t)lds_dst;          // addr[31:0] = LDS byte address
  v4u g0;
  g0[0] = 1u;                                           // count=1 valid user descriptor
  g0[1] = lds;                                          // lds_addr
  g0[2] = (unsigned)ga;                                 // global_addr[31:0]
  g0[3] = (unsigned)((ga >> 32) & 0x01FFFFFFull) | (2u << 30);  // addr[56:32] | type=2
  v8i g1;
  // data_size=2B (1<<16) | pad_enable (1<<20) | pad_interval=32DW (4<<22) | pad_amount=4DW (3<<25)
  g1[0] = (int)((1u << 16) | (1u << 20) | (4u << 22) | (3u << 25));
  g1[1] = (int)(64u << 16);                             // tensor_dim0 = 64
  g1[2] = (int)(((unsigned)rows & 0xFFFFu) << 16);      // tensor_dim1 = rows
  g1[3] = (int)(64u << 16);                             // tile_dim0 = 64
  g1[4] = rows;                                         // tile_dim1 = rows (tile_dim2 = 0)
  g1[5] = 64;                                           // tensor_dim0_stride = 64
  g1[6] = 0;
  g1[7] = 0;
  v4i z4 = {0, 0, 0, 0};                                // groups 2/3 unused (2D tensor)
#if __clang_major__ >= 23
  v8i z8 = {0, 0, 0, 0, 0, 0, 0, 0};
  __builtin_amdgcn_tensor_load_to_lds(g0, g1, z4, z4, z8, 0);
#else
  __builtin_amdgcn_tensor_load_to_lds(g0, g1, z4, z4, 0);
#endif
  __builtin_amdgcn_s_wait_tensorcnt(0);
#else
  (void)gsrc; (void)lds_dst; (void)rows;
#endif
}

// Vectorized fallback tile copy (bf16, pitch 64 -> 72), all 256 threads.
static __device__ __forceinline__ void copy_tile_bf16(const bf16_t* __restrict__ g,
                                                      bf16_t* __restrict__ lds, int rows) {
  for (int i = (int)threadIdx.x; i < rows * 8; i += 256) {
    int rr = i >> 3, c8 = (i & 7) << 3;
    *(v4u*)&lds[rr * LDP + c8] = *(const v4u*)(g + (size_t)rr * 64 + c8);
  }
}

// ---------------------------------------------------------------------------
// 1) QKV projection: C[4096x1024] = X[4096x1024] @ W[1024x1024]^T + bias,
//    scatter-stored to head layout [b=n*16+h][l][dd] as bf16. which: 0=q,1=k,2=v.
// ---------------------------------------------------------------------------
__global__ void __launch_bounds__(256)
qkv_proj_kernel(const float* __restrict__ Xq, const float* __restrict__ Xk,
                const float* __restrict__ Xv, const float* __restrict__ Wall,
                const float* __restrict__ Ball,
                bf16_t* __restrict__ qh, bf16_t* __restrict__ kh, bf16_t* __restrict__ vh) {
  const int j0 = (int)blockIdx.x * 64;            // output-feature tile
  const int r0 = (int)blockIdx.y * 64;            // row tile over L*N = 4096
  const int which = (int)blockIdx.z;              // 0=q 1=k 2=v
  const float* X = (which == 0) ? Xq : (which == 1) ? Xk : Xv;
  const float* W = Wall + (size_t)which * 1024 * 1024;
  const float* B = Ball + which * 1024;
  bf16_t* out = (which == 0) ? qh : (which == 1) ? kh : vh;

  __shared__ __align__(16) bf16_t As[64 * LDP];
  __shared__ __align__(16) bf16_t Ws[64 * LDP];

  const int wv = (int)(threadIdx.x >> 5), lane = (int)(threadIdx.x & 31u);
  const int wm = wv & 3, wn = wv >> 2;            // 4x2 wave grid over 64x64 tile
  const int m16 = wm * 16;
  const int half = lane >> 4;

  v8f acc[2] = { vzero8(), vzero8() };

  for (int k0 = 0; k0 < 1024; k0 += 64) {
    __syncthreads();
    // b128 loads, f32->bf16 pack, b64 LDS stores (4 chunks per thread).
    for (int i = (int)threadIdx.x; i < 64 * 16; i += 256) {
      int rr = i >> 4, c4 = (i & 15) << 2;
      v4f xa = *(const v4f*)(X + (size_t)(r0 + rr) * 1024 + k0 + c4);
      v4f wa = *(const v4f*)(W + (size_t)(j0 + rr) * 1024 + k0 + c4);
      v2u xp = { pk2(xa[0], xa[1]), pk2(xa[2], xa[3]) };
      v2u wp = { pk2(wa[0], wa[1]), pk2(wa[2], wa[3]) };
      *(v2u*)&As[rr * LDP + c4] = xp;
      *(v2u*)&Ws[rr * LDP + c4] = wp;
    }
    __syncthreads();
#pragma unroll
    for (int kc = 0; kc < 64; kc += 32) {
      v16bf a = ldfrag(As + m16 * LDP, LDP, kc);
#pragma unroll
      for (int nt = 0; nt < 2; ++nt) {
        v16bf b = ldfrag(Ws + (wn * 32 + nt * 16) * LDP, LDP, kc);
        acc[nt] = wmma_bf16(a, b, acc[nt]);
      }
    }
  }

  const float scale = (which == 0) ? 0.125f : 1.0f;   // fold 1/sqrt(64) into Q
#pragma unroll
  for (int nt = 0; nt < 2; ++nt) {
    int j = j0 + wn * 32 + nt * 16 + (lane & 15);
    float bj = B[j];
    int h = (j >> 6) & 15, dd = j & 63;
#pragma unroll
    for (int g = 0; g < 8; ++g) {
      int r = r0 + m16 + half * 8 + g;                // row in (L,N) flattening
      int l = r >> 1, n = r & 1;
      float val = (acc[nt][g] + bj) * scale;
      out[((size_t)(n * 16 + h) * 2048 + l) * 64 + dd] = f2bf(val);
    }
  }
}

// ---------------------------------------------------------------------------
// 2) Flash stats pass: per (b,l) compute rowmax m and sumexp over all S.
//    Block = 128 query rows x all 2048 keys; wave owns a 16-row stripe.
// ---------------------------------------------------------------------------
__global__ void __launch_bounds__(256)
attn_stats_kernel(const bf16_t* __restrict__ qh, const bf16_t* __restrict__ kh,
                  float* __restrict__ stats_m, float* __restrict__ stats_s) {
  const int l0 = (int)blockIdx.x * 128;
  const int b  = (int)blockIdx.y;

  __shared__ __align__(16) bf16_t Qs[128 * LDP];
  __shared__ __align__(16) bf16_t Ks[64 * LDP];

  const int wv = (int)(threadIdx.x >> 5), lane = (int)(threadIdx.x & 31u);
  const int m16 = wv * 16, half = lane >> 4;

  const bf16_t* Qb = qh + ((size_t)b * 2048 + l0) * 64;
#if HAVE_TDM
  if (wv == 0) tdm_load_tile_bf16(Qb, Qs, 128);
#else
  copy_tile_bf16(Qb, Qs, 128);
#endif

  float mrun[8], srun[8];
#pragma unroll
  for (int g = 0; g < 8; ++g) { mrun[g] = -3.0e38f; srun[g] = 0.f; }

  for (int s0 = 0; s0 < 2048; s0 += 64) {
    __syncthreads();
    const bf16_t* Kb = kh + ((size_t)b * 2048 + s0) * 64;
    if (s0 + 64 < 2048)   // global_prefetch_b8 of the next K tile into L2
      __builtin_prefetch((const void*)(Kb + 64 * 64 + (threadIdx.x << 4)), 0, 0);
#if HAVE_TDM
    if (wv == 0) tdm_load_tile_bf16(Kb, Ks, 64);
#else
    copy_tile_bf16(Kb, Ks, 64);
#endif
    __syncthreads();

    v8f acc[4] = { vzero8(), vzero8(), vzero8(), vzero8() };
#pragma unroll
    for (int kc = 0; kc < 64; kc += 32) {
      v16bf a = ldfrag(Qs + m16 * LDP, LDP, kc);
#pragma unroll
      for (int ni = 0; ni < 4; ++ni) {
        v16bf bb = ldfrag(Ks + ni * 16 * LDP, LDP, kc);
        acc[ni] = wmma_bf16(a, bb, acc[ni]);
      }
    }

    // Online (m, sumexp) update.  Row g lives in VGPR g; columns are spread
    // over ni tiles and the 16 lanes of each half-wave -> shfl_xor reduce.
#pragma unroll
    for (int g = 0; g < 8; ++g) {
      float mloc = fmaxf(fmaxf(acc[0][g], acc[1][g]), fmaxf(acc[2][g], acc[3][g]));
#pragma unroll
      for (int ofs = 8; ofs > 0; ofs >>= 1) mloc = fmaxf(mloc, __shfl_xor(mloc, ofs, 32));
      float mnew = fmaxf(mrun[g], mloc);
      float e = __expf(acc[0][g] - mnew) + __expf(acc[1][g] - mnew) +
                __expf(acc[2][g] - mnew) + __expf(acc[3][g] - mnew);
#pragma unroll
      for (int ofs = 8; ofs > 0; ofs >>= 1) e += __shfl_xor(e, ofs, 32);
      srun[g] = srun[g] * __expf(mrun[g] - mnew) + e;
      mrun[g] = mnew;
    }
  }

  if ((lane & 15) == 0) {
#pragma unroll
    for (int g = 0; g < 8; ++g) {
      int row = l0 + m16 + half * 8 + g;
      stats_m[(size_t)b * 2048 + row] = mrun[g];
      stats_s[(size_t)b * 2048 + row] = srun[g];
    }
  }
}

// ---------------------------------------------------------------------------
// 3) avg_uncertainty[n][l] = mean_h S / (sumexp + S + 1e-10)
// ---------------------------------------------------------------------------
__global__ void __launch_bounds__(256)
uncertainty_kernel(const float* __restrict__ stats_s, float* __restrict__ unc) {
  int idx = (int)(blockIdx.x * 256 + threadIdx.x);
  if (idx >= 4096) return;
  int n = idx >> 11, l = idx & 2047;
  float acc = 0.f;
#pragma unroll
  for (int h = 0; h < 16; ++h)
    acc += 2048.0f / (stats_s[((size_t)(n * 16 + h)) * 2048 + l] + 2048.0f + 1e-10f);
  unc[idx] = acc * (1.0f / 16.0f);
}

// ---------------------------------------------------------------------------
// 4) Recompute scores, emit exact weights (537MB, the HBM-bound part),
//    and fuse P@V.  P goes through a per-wave LDS staging tile to convert
//    the WMMA C-layout into an A-fragment layout.
// ---------------------------------------------------------------------------
__global__ void __launch_bounds__(256)
attn_weights_pv_kernel(const bf16_t* __restrict__ qh, const bf16_t* __restrict__ kh,
                       const bf16_t* __restrict__ vh,
                       const float* __restrict__ stats_m, const float* __restrict__ stats_s,
                       float* __restrict__ w_out, bf16_t* __restrict__ aout) {
  const int l0 = (int)blockIdx.x * 128;
  const int b  = (int)blockIdx.y;

  __shared__ __align__(16) bf16_t Qs[128 * LDP];
  __shared__ __align__(16) bf16_t Ks[64 * LDP];
  __shared__ __align__(16) bf16_t Vt[64 * LDP];        // V transposed: [d][s]
  __shared__ __align__(16) bf16_t Ps[8 * 16 * LDP];    // per-wave P staging [16][64]

  const int wv = (int)(threadIdx.x >> 5), lane = (int)(threadIdx.x & 31u);
  const int m16 = wv * 16, half = lane >> 4;
  bf16_t* Pw = Ps + wv * 16 * LDP;

  const bf16_t* Qb = qh + ((size_t)b * 2048 + l0) * 64;
#if HAVE_TDM
  if (wv == 0) tdm_load_tile_bf16(Qb, Qs, 128);
#else
  copy_tile_bf16(Qb, Qs, 128);
#endif

  float mr[8], inv[8];
#pragma unroll
  for (int g = 0; g < 8; ++g) {
    int row = l0 + m16 + half * 8 + g;
    mr[g]  = stats_m[(size_t)b * 2048 + row];
    inv[g] = 1.0f / (stats_s[(size_t)b * 2048 + row] + 2048.0f + 1e-10f);
  }

  v8f oacc[4] = { vzero8(), vzero8(), vzero8(), vzero8() };

  for (int s0 = 0; s0 < 2048; s0 += 64) {
    __syncthreads();
    const bf16_t* Kb = kh + ((size_t)b * 2048 + s0) * 64;
    const bf16_t* Vb = vh + ((size_t)b * 2048 + s0) * 64;
#if HAVE_TDM
    if (wv == 0) tdm_load_tile_bf16(Kb, Ks, 64);
#else
    copy_tile_bf16(Kb, Ks, 64);
#endif
    // V transpose into LDS: b128 global reads, scalar transposed LDS writes.
    for (int i = (int)threadIdx.x; i < 64 * 8; i += 256) {
      int ss = i >> 3, d0 = (i & 7) << 3;
      union { v4u v; bf16_t e[8]; } u;
      u.v = *(const v4u*)(Vb + (size_t)ss * 64 + d0);
#pragma unroll
      for (int j = 0; j < 8; ++j) Vt[(d0 + j) * LDP + ss] = u.e[j];
    }
    __syncthreads();

    // scores for this 128x64 block (8 WMMAs per wave)
    v8f acc[4] = { vzero8(), vzero8(), vzero8(), vzero8() };
#pragma unroll
    for (int kc = 0; kc < 64; kc += 32) {
      v16bf a = ldfrag(Qs + m16 * LDP, LDP, kc);
#pragma unroll
      for (int ni = 0; ni < 4; ++ni) {
        v16bf bb = ldfrag(Ks + ni * 16 * LDP, LDP, kc);
        acc[ni] = wmma_bf16(a, bb, acc[ni]);
      }
    }

    // credal weights: w = (exp(s-m)+1) / (sumexp + S + 1e-10); store f32 to
    // d_out (mandatory traffic) and bf16 to per-wave LDS staging for P@V.
#pragma unroll
    for (int ni = 0; ni < 4; ++ni) {
#pragma unroll
      for (int g = 0; g < 8; ++g) {
        float wt = (__expf(acc[ni][g] - mr[g]) + 1.0f) * inv[g];
        int row = l0 + m16 + half * 8 + g;
        int col = s0 + ni * 16 + (lane & 15);
        w_out[((size_t)b * 2048 + row) * 2048 + col] = wt;
        Pw[(half * 8 + g) * LDP + ni * 16 + (lane & 15)] = f2bf(wt);
      }
    }
    // Own-wave LDS store->load ordering (DS ops are in-order per wave; make
    // the dependency explicit for the scheduler).
    asm volatile("s_wait_dscnt 0" ::: "memory");

    // P(16x64) @ V(64x64): A frags from Pw, B frags from transposed V tile.
#pragma unroll
    for (int kc = 0; kc < 64; kc += 32) {
      v16bf pa = ldfrag(Pw, LDP, kc);
#pragma unroll
      for (int ni = 0; ni < 4; ++ni) {
        v16bf vb = ldfrag(Vt + ni * 16 * LDP, LDP, kc);
        oacc[ni] = wmma_bf16(pa, vb, oacc[ni]);
      }
    }
  }

  const int n = b >> 4, h = b & 15;
#pragma unroll
  for (int ni = 0; ni < 4; ++ni) {
#pragma unroll
    for (int g = 0; g < 8; ++g) {
      int row = l0 + m16 + half * 8 + g;
      int r = row * 2 + n;                             // (L,N) flattening
      int col = h * 64 + ni * 16 + (lane & 15);
      aout[(size_t)r * 1024 + col] = f2bf(oacc[ni][g]);
    }
  }
}

// ---------------------------------------------------------------------------
// 5) Output projection: out[4096x1024] = aout(bf16) @ Wo^T + bo  (f32 store)
// ---------------------------------------------------------------------------
__global__ void __launch_bounds__(256)
out_proj_kernel(const bf16_t* __restrict__ A, const float* __restrict__ Wo,
                const float* __restrict__ bo, float* __restrict__ out) {
  const int j0 = (int)blockIdx.x * 64;
  const int r0 = (int)blockIdx.y * 64;

  __shared__ __align__(16) bf16_t As[64 * LDP];
  __shared__ __align__(16) bf16_t Ws[64 * LDP];

  const int wv = (int)(threadIdx.x >> 5), lane = (int)(threadIdx.x & 31u);
  const int wm = wv & 3, wn = wv >> 2;
  const int m16 = wm * 16, half = lane >> 4;

  v8f acc[2] = { vzero8(), vzero8() };

  for (int k0 = 0; k0 < 1024; k0 += 64) {
    __syncthreads();
    for (int i = (int)threadIdx.x; i < 64 * 8; i += 256) {
      int rr = i >> 3, c8 = (i & 7) << 3;
      // A tile: straight b128 bf16 copy
      *(v4u*)&As[rr * LDP + c8] = *(const v4u*)(A + (size_t)(r0 + rr) * 1024 + k0 + c8);
      // W tile: two b128 f32 loads -> packed b128 bf16 store
      v4f w0 = *(const v4f*)(Wo + (size_t)(j0 + rr) * 1024 + k0 + c8);
      v4f w1 = *(const v4f*)(Wo + (size_t)(j0 + rr) * 1024 + k0 + c8 + 4);
      v4u wp = { pk2(w0[0], w0[1]), pk2(w0[2], w0[3]), pk2(w1[0], w1[1]), pk2(w1[2], w1[3]) };
      *(v4u*)&Ws[rr * LDP + c8] = wp;
    }
    __syncthreads();
#pragma unroll
    for (int kc = 0; kc < 64; kc += 32) {
      v16bf a = ldfrag(As + m16 * LDP, LDP, kc);
#pragma unroll
      for (int nt = 0; nt < 2; ++nt) {
        v16bf b = ldfrag(Ws + (wn * 32 + nt * 16) * LDP, LDP, kc);
        acc[nt] = wmma_bf16(a, b, acc[nt]);
      }
    }
  }

#pragma unroll
  for (int nt = 0; nt < 2; ++nt) {
    int j = j0 + wn * 32 + nt * 16 + (lane & 15);
    float bj = bo[j];
#pragma unroll
    for (int g = 0; g < 8; ++g) {
      int r = r0 + m16 + half * 8 + g;
      out[(size_t)r * 1024 + j] = acc[nt][g] + bj;
    }
  }
}

// ---------------------------------------------------------------------------
// Host-side launch.  Workspace layout (~33 MB):
//   qh/kh/vh : bf16 [32][2048][64]  (8 MB each)
//   stats_m/stats_s : f32 [32][2048]
//   aout : bf16 [4096][1024] (8 MB)
// d_out: [attn_output 4,194,304 f32][attn_weights 134,217,728 f32][unc 4096 f32]
// ---------------------------------------------------------------------------
extern "C" void kernel_launch(void* const* d_in, const int* in_sizes, int n_in,
                              void* d_out, int out_size, void* d_ws, size_t ws_size,
                              hipStream_t stream) {
  (void)in_sizes; (void)n_in; (void)out_size; (void)ws_size;
  const float* query = (const float*)d_in[0];
  const float* key   = (const float*)d_in[1];
  const float* value = (const float*)d_in[2];
  const float* ipw   = (const float*)d_in[3];
  const float* ipb   = (const float*)d_in[4];
  const float* opw   = (const float*)d_in[5];
  const float* opb   = (const float*)d_in[6];

  float* out       = (float*)d_out;
  float* attn_out  = out;
  float* weights   = out + (size_t)4194304;
  float* unc       = weights + (size_t)134217728;

  const size_t headElems = (size_t)32 * 2048 * 64;     // 4,194,304 bf16
  bf16_t* qh = (bf16_t*)d_ws;
  bf16_t* kh = qh + headElems;
  bf16_t* vh = kh + headElems;
  float* stats_m = (float*)(vh + headElems);
  float* stats_s = stats_m + (size_t)32 * 2048;
  bf16_t* aout = (bf16_t*)(stats_s + (size_t)32 * 2048);

  dim3 blk(256, 1, 1);
  qkv_proj_kernel<<<dim3(16, 64, 3), blk, 0, stream>>>(query, key, value, ipw, ipb, qh, kh, vh);
  attn_stats_kernel<<<dim3(16, 32, 1), blk, 0, stream>>>(qh, kh, stats_m, stats_s);
  uncertainty_kernel<<<dim3(16, 1, 1), blk, 0, stream>>>(stats_s, unc);
  attn_weights_pv_kernel<<<dim3(16, 32, 1), blk, 0, stream>>>(qh, kh, vh, stats_m, stats_s,
                                                              weights, aout);
  out_proj_kernel<<<dim3(16, 64, 1), blk, 0, stream>>>(aout, opw, opb, attn_out);
}
